// QcRelPositionMultiheadAttention_1357209666477
// MI455X (gfx1250) — compile-verified, matrix operand order checked by reading, not run on hardware
//
#include <hip/hip_runtime.h>
#include <hip/hip_bf16.h>

// ---------------------------------------------------------------------------
// QcRelPositionMultiheadAttention for MI455X (gfx1250), wave32 + WMMA bf16.
// All contractions run through v_wmma_f32_16x16x32_bf16 (fp32 accum).
// ---------------------------------------------------------------------------

typedef __bf16 v16bf __attribute__((ext_vector_type(16)));
typedef __bf16 v4bf  __attribute__((ext_vector_type(4)));
typedef float  v8f   __attribute__((ext_vector_type(8)));

// Problem constants (fixed by the reference)
constexpr int kS   = 512;   // query sequence
constexpr int kB   = 16;    // batch
constexpr int kE   = 512;   // embed
constexpr int kA   = 512;   // attention dim
constexpr int kH   = 8;     // heads
constexpr int kP   = 4;     // pos dim
constexpr int kL   = 512;   // left context
constexpr int kKV  = 1024;  // kv length = L + S
constexpr int kHD  = 64;    // head dim
constexpr int kVD  = 32;    // value head dim
constexpr int kIO  = 1312;  // in_proj out channels (2A + A/2 + H*P)
constexpr int kS2  = 1535;  // pos_emb length

// ---------------------------------------------------------------------------
// Helpers
// ---------------------------------------------------------------------------
// Native cast: lets clang use gfx1250's hardware bf16 conversion
// (v_cvt_pk_bf16_f32 class) instead of a software RNE bit-twiddle.
__device__ __forceinline__ __bf16 f2bf(float f) { return (__bf16)f; }

__device__ __forceinline__ v4bf f2bf4(float4 v) {
  v4bf p = { f2bf(v.x), f2bf(v.y), f2bf(v.z), f2bf(v.w) };
  return p;
}

__device__ __forceinline__ v8f wmma_bf16(v16bf a, v16bf b, v8f c) {
  // (neg_a, A, neg_b, B, c_mod, C, reuse_a, reuse_b)
  return __builtin_amdgcn_wmma_f32_16x16x32_bf16(false, a, false, b, (short)0, c,
                                                 false, false);
}

// A/B fragment (16x32 bf16): lane&15 selects the M (or N) row, the lane's hi bit
// selects K base 0/8; each lane holds K = kb..kb+7 and kb+16..kb+23 (two
// contiguous 16B runs -> b128 loads).
__device__ __forceinline__ v16bf load_ab_frag(const __bf16* base, int stride) {
  int lane = threadIdx.x & 31;
  int m    = lane & 15;
  int kb   = (lane >> 4) << 3;
  const __bf16* p = base + (size_t)m * stride + kb;
  v16bf r;
#pragma unroll
  for (int i = 0; i < 8; ++i) r[i] = p[i];
#pragma unroll
  for (int i = 0; i < 8; ++i) r[8 + i] = p[16 + i];
  return r;
}

// Same fragment but sourced from fp32 (converted per element).
__device__ __forceinline__ v16bf load_ab_frag_f32(const float* base, int stride) {
  int lane = threadIdx.x & 31;
  int m    = lane & 15;
  int kb   = (lane >> 4) << 3;
  const float* p = base + (size_t)m * stride + kb;
  v16bf r;
#pragma unroll
  for (int i = 0; i < 8; ++i) r[i] = f2bf(p[i]);
#pragma unroll
  for (int i = 0; i < 8; ++i) r[8 + i] = f2bf(p[16 + i]);
  return r;
}

// ---------------------------------------------------------------------------
// Generic GEMM: C[M,N] = A[M,K] * Bt[N,K]^T (+bias), fp32 in/out, bf16 WMMA.
// Block = 256 threads (8 waves). 128x64 C tile; each wave owns a 32x32
// register tile (2x2 fragments -> 4 WMMAs per k-step from 4 fragment loads).
// Bounds handling is branchless: clamped-row b128 loads + zero-select, so the
// fill path is identical for interior and edge tiles (no serialized loops).
// ---------------------------------------------------------------------------
__global__ void __launch_bounds__(256)
gemm_bf16_wmma(const float* __restrict__ A, const float* __restrict__ Bt,
               const float* __restrict__ bias, float* __restrict__ C,
               int M, int N, int K) {
  __shared__ alignas(16) __bf16 As[128][32];  // 8 KB
  __shared__ alignas(16) __bf16 Bs[64][32];   // 4 KB

  const int bm   = blockIdx.y * 128;
  const int bn   = blockIdx.x * 64;
  const int tid  = threadIdx.x;
  const int wave = tid >> 5;
  const int lane = tid & 31;
  const int m0   = (wave & 3) * 32;   // wave's 32-row sub-tile
  const int n0   = (wave >> 2) * 32;  // wave's 32-col sub-tile
  const v4bf zero4 = {};

  v8f acc00 = {}, acc01 = {}, acc10 = {}, acc11 = {};

  for (int k0 = 0; k0 < K; k0 += 32) {
    // ---- fill As (128x32 = 1024 vec4) ----
#pragma unroll
    for (int i = tid; i < 1024; i += 256) {
      int r = i >> 3, c = (i & 7) * 4;
      int gr = bm + r;
      int cr = gr < M ? gr : M - 1;
      float4 va = *(const float4*)&A[(size_t)cr * K + k0 + c];
      *(v4bf*)&As[r][c] = (gr < M) ? f2bf4(va) : zero4;
    }
    // ---- fill Bs (64x32 = 512 vec4) ----
#pragma unroll
    for (int i = tid; i < 512; i += 256) {
      int r = i >> 3, c = (i & 7) * 4;
      int gr = bn + r;
      int cr = gr < N ? gr : N - 1;
      float4 vb = *(const float4*)&Bt[(size_t)cr * K + k0 + c];
      *(v4bf*)&Bs[r][c] = (gr < N) ? f2bf4(vb) : zero4;
    }
    if (k0 + 32 < K) {  // prefetch next K tile (emits global_prefetch_b8)
      int pr = bm + (tid >> 1);
      __builtin_prefetch(&A[(size_t)(pr < M ? pr : M - 1) * K + k0 + 32], 0, 1);
      int qr = bn + (tid >> 2);
      __builtin_prefetch(&Bt[(size_t)(qr < N ? qr : N - 1) * K + k0 + 32], 0, 1);
    }
    __syncthreads();

    v16bf af0 = load_ab_frag(&As[m0][0], 32);
    v16bf af1 = load_ab_frag(&As[m0 + 16][0], 32);
    v16bf bf0 = load_ab_frag(&Bs[n0][0], 32);
    v16bf bf1 = load_ab_frag(&Bs[n0 + 16][0], 32);
    acc00 = wmma_bf16(af0, bf0, acc00);
    acc01 = wmma_bf16(af0, bf1, acc01);
    acc10 = wmma_bf16(af1, bf0, acc10);
    acc11 = wmma_bf16(af1, bf1, acc11);
    __syncthreads();
  }

  const int hi  = lane >> 4;
  const int col = lane & 15;
  const int gn0 = bn + n0 + col;
  const int gn1 = gn0 + 16;
  const float b0 = (bias && gn0 < N) ? bias[gn0] : 0.f;
  const float b1 = (bias && gn1 < N) ? bias[gn1] : 0.f;
#pragma unroll
  for (int r = 0; r < 8; ++r) {
    int gmA = bm + m0 + r + 8 * hi;        // first m sub-tile
    int gmB = gmA + 16;                     // second m sub-tile
    if (gmA < M) {
      if (gn0 < N) C[(size_t)gmA * N + gn0] = acc00[r] + b0;
      if (gn1 < N) C[(size_t)gmA * N + gn1] = acc01[r] + b1;
    }
    if (gmB < M) {
      if (gn0 < N) C[(size_t)gmB * N + gn0] = acc10[r] + b0;
      if (gn1 < N) C[(size_t)gmB * N + gn1] = acc11[r] + b1;
    }
  }
}

// ---------------------------------------------------------------------------
// Packing kernels (vectorized): split xp [S,B,1312] into head-major bf16 Q/K,
// n-contiguous transposed bf16 Vt, fp32 P, and emit cached_{key,val}_out.
// ---------------------------------------------------------------------------
__global__ void pack_q_kernel(const float* __restrict__ xp, __bf16* __restrict__ Qb) {
  size_t i = (size_t)blockIdx.x * 256 + threadIdx.x;   // vec4 over B*H*S*HD
  if (i >= (size_t)kB * kH * kS * kHD / 4) return;
  int d = (i % (kHD / 4)) * 4; size_t t = i / (kHD / 4);
  int s = t % kS;  t /= kS;
  int h = t % kH;  int b = t / kH;
  float4 v = *(const float4*)&xp[((size_t)s * kB + b) * kIO + h * kHD + d];
  *(v4bf*)&Qb[i * 4] = f2bf4(v);
}

__global__ void pack_k_kernel(const float* __restrict__ xp,
                              const float* __restrict__ cached_key,
                              __bf16* __restrict__ Kb) {
  size_t i = (size_t)blockIdx.x * 256 + threadIdx.x;   // vec4 over B*H*KV*HD
  if (i >= (size_t)kB * kH * kKV * kHD / 4) return;
  int d = (i % (kHD / 4)) * 4; size_t t = i / (kHD / 4);
  int n = t % kKV; t /= kKV;
  int h = t % kH;  int b = t / kH;
  float4 v;
  if (n < kL) v = *(const float4*)&cached_key[((size_t)n * kB + b) * kA + h * kHD + d];
  else        v = *(const float4*)&xp[((size_t)(n - kL) * kB + b) * kIO + kA + h * kHD + d];
  *(v4bf*)&Kb[i * 4] = f2bf4(v);
}

// V is stored transposed: Vt[b][h][d][n] (n contiguous) so the AV B-fragments
// are two contiguous 16B loads. LDS-transposing tile kernel, 64 n x 32 d.
__global__ void __launch_bounds__(256)
pack_v_kernel(const float* __restrict__ xp, const float* __restrict__ cached_val,
              __bf16* __restrict__ Vt) {
  __shared__ alignas(16) __bf16 tile[kVD][64 + 8];
  const int nt = blockIdx.x, h = blockIdx.y, b = blockIdx.z;
  const int tid = threadIdx.x;
  const int n0 = nt * 64;
  for (int i = tid; i < 512; i += 256) {   // 64 rows x 8 vec4
    int nl = i >> 3, d4 = (i & 7) * 4;
    int n = n0 + nl;
    float4 v = (n < kL)
      ? *(const float4*)&cached_val[((size_t)n * kB + b) * (kA / 2) + h * kVD + d4]
      : *(const float4*)&xp[((size_t)(n - kL) * kB + b) * kIO + 2 * kA + h * kVD + d4];
    tile[d4 + 0][nl] = f2bf(v.x);
    tile[d4 + 1][nl] = f2bf(v.y);
    tile[d4 + 2][nl] = f2bf(v.z);
    tile[d4 + 3][nl] = f2bf(v.w);
  }
  __syncthreads();
  for (int i = tid; i < kVD * 8; i += 256) {  // 32 d-rows x 8 chunks of 8
    int d = i >> 3, c = (i & 7) * 8;
    __bf16* dst = &Vt[(((size_t)b * kH + h) * kVD + d) * kKV + n0 + c];
    ((v4bf*)dst)[0] = *(v4bf*)&tile[d][c];
    ((v4bf*)dst)[1] = *(v4bf*)&tile[d][c + 4];
  }
}

__global__ void pack_p_kernel(const float* __restrict__ xp, float* __restrict__ Pp) {
  size_t i = (size_t)blockIdx.x * 256 + threadIdx.x;   // vec4 over B*H*S
  if (i >= (size_t)kB * kH * kS) return;
  int s = i % kS; size_t t = i / kS;
  int h = t % kH; int b = t / kH;
  *(float4*)&Pp[i * 4] =
      *(const float4*)&xp[((size_t)s * kB + b) * kIO + 2 * kA + kA / 2 + h * kP];
}

__global__ void cache_out_kernel(const float* __restrict__ xp,
                                 float* __restrict__ cko,
                                 float* __restrict__ cvo) {
  size_t nk4 = (size_t)kS * kB * kA / 4;
  size_t nv4 = (size_t)kS * kB * (kA / 2) / 4;
  size_t i = (size_t)blockIdx.x * 256 + threadIdx.x;
  if (i < nk4) {
    int c = (i % (kA / 4)) * 4; size_t t = i / (kA / 4);
    int b = t % kB; int s = t / kB;
    *(float4*)&cko[i * 4] = *(const float4*)&xp[((size_t)s * kB + b) * kIO + kA + c];
  } else if (i < nk4 + nv4) {
    size_t j = i - nk4;
    int c = (j % (kA / 8)) * 4; size_t t = j / (kA / 8);
    int b = t % kB; int s = t / kB;
    *(float4*)&cvo[j * 4] = *(const float4*)&xp[((size_t)s * kB + b) * kIO + 2 * kA + c];
  }
}

// ---------------------------------------------------------------------------
// Attention: one block per (b, h, 16-row s tile). 256 threads = 8 waves.
// QK^T via WMMA (+rel-pos bias with the take_along_axis gather folded into
// addressing), fp32 softmax over 1024 cols in LDS, attn*V via WMMA.
// Weights (268 MB stream) written with non-temporal stores. LDS ~69 KB.
// ---------------------------------------------------------------------------
__global__ void __launch_bounds__(256)
attn_kernel(const __bf16* __restrict__ Qb, const __bf16* __restrict__ Kb,
            const __bf16* __restrict__ Vt, const float* __restrict__ Pp,
            const float* __restrict__ posm, float* __restrict__ weights_out,
            float* __restrict__ attnout) {
  __shared__ alignas(16) __bf16 Qs[16][kHD];   // 2 KB
  __shared__ float  scores[16][kKV];           // 64 KB
  __shared__ float  Ppl[16][kP];               // 256 B
  __shared__ float  red[16][16];               // 1 KB
  __shared__ float  outacc[16][kVD];           // 2 KB

  const int stile = blockIdx.x, h = blockIdx.y, b = blockIdx.z;
  const int s0   = stile * 16;
  const int tid  = threadIdx.x;
  const int wave = tid >> 5;
  const int lane = tid & 31;
  const int bh   = b * kH + h;

  // Stage Q tile (16B copies) + per-row pos projections; zero the accumulator.
  const __bf16* Qg = Qb + ((size_t)bh * kS + s0) * kHD;
  if (tid < 128) ((int4*)Qs)[tid] = ((const int4*)Qg)[tid];
  if (tid < 16 * kP)
    Ppl[tid >> 2][tid & 3] = Pp[((size_t)bh * kS + s0 + (tid >> 2)) * kP + (tid & 3)];
  for (int i = tid; i < 16 * kVD; i += 256) ((float*)outacc)[i] = 0.f;
  __syncthreads();

  v16bf qf0 = load_ab_frag(&Qs[0][0], kHD);   // d = 0..31
  v16bf qf1 = load_ab_frag(&Qs[0][32], kHD);  // d = 32..63

  // ---- scores = Q*K^T + pos_weights (gathered) ----
  const __bf16* Kg = Kb + (size_t)bh * kKV * kHD;
  const int hi = lane >> 4, col = lane & 15;
  for (int nt = 0; nt < 8; ++nt) {
    int n0 = wave * 128 + nt * 16;
    v16bf kf0 = load_ab_frag(Kg + (size_t)n0 * kHD, kHD);
    v16bf kf1 = load_ab_frag(Kg + (size_t)n0 * kHD + 32, kHD);
    v8f acc = {};
    acc = wmma_bf16(qf0, kf0, acc);
    acc = wmma_bf16(qf1, kf1, acc);
#pragma unroll
    for (int r = 0; r < 8; ++r) {
      int sl = r + 8 * hi;
      int j  = n0 + col;
      int s  = s0 + sl;
      int idx = (kS - 1) - s + j;                    // take_along_axis gather
      float4 pr = *(const float4*)&posm[((size_t)idx * kH + h) * kP];
      float pw = Ppl[sl][0] * pr.x + Ppl[sl][1] * pr.y +
                 Ppl[sl][2] * pr.z + Ppl[sl][3] * pr.w;
      scores[sl][j] = acc[r] + pw;
    }
  }
  __syncthreads();

  // ---- softmax over j (16 threads per row) ----
  {
    int row = tid >> 4, c = tid & 15;
    float mx = -3.4e38f;
    for (int j = c; j < kKV; j += 16) mx = fmaxf(mx, scores[row][j]);
    red[row][c] = mx;
    __syncthreads();
    mx = red[row][0];
#pragma unroll
    for (int i = 1; i < 16; ++i) mx = fmaxf(mx, red[row][i]);
    __syncthreads();
    float sum = 0.f;
    for (int j = c; j < kKV; j += 16) {
      float e = __expf(scores[row][j] - mx);
      scores[row][j] = e;
      sum += e;
    }
    red[row][c] = sum;
    __syncthreads();
    sum = 0.f;
#pragma unroll
    for (int i = 0; i < 16; ++i) sum += red[row][i];
    float inv = 1.f / sum;
    float* wrow = weights_out + ((size_t)bh * kS + s0 + row) * kKV;
    for (int j = c; j < kKV; j += 16) {
      float pv = scores[row][j] * inv;
      scores[row][j] = pv;
      __builtin_nontemporal_store(pv, &wrow[j]);   // bypass L2 for 268MB stream
    }
  }
  __syncthreads();

  // ---- out = attn * V (WMMA; Vt is n-contiguous -> b128 fragment loads) ----
  const __bf16* Vg = Vt + (size_t)bh * kVD * kKV;
  v8f o0 = {}, o1 = {};
  for (int nc = 0; nc < 4; ++nc) {
    int n0 = wave * 128 + nc * 32;
    v16bf af  = load_ab_frag_f32(&scores[0][n0], kKV);
    v16bf vf0 = load_ab_frag(Vg + n0, kKV);                     // d = 0..15
    v16bf vf1 = load_ab_frag(Vg + (size_t)16 * kKV + n0, kKV);  // d = 16..31
    o0 = wmma_bf16(af, vf0, o0);
    o1 = wmma_bf16(af, vf1, o1);
  }
#pragma unroll
  for (int r = 0; r < 8; ++r) {
    int sl = r + 8 * hi;
    atomicAdd(&outacc[sl][col], o0[r]);
    atomicAdd(&outacc[sl][col + 16], o1[r]);
  }
  __syncthreads();

  for (int i = tid; i < 16 * kVD; i += 256) {
    int sl = i >> 5, d = i & 31;
    attnout[((size_t)(s0 + sl) * kB + b) * (kA / 2) + h * kVD + d] = outacc[sl][d];
  }
}

// ---------------------------------------------------------------------------
// Launch. d_out = [out | weights | cached_key_out | cached_val_out] fp32.
// Workspace usage ~86 MB.
// ---------------------------------------------------------------------------
extern "C" void kernel_launch(void* const* d_in, const int* in_sizes, int n_in,
                              void* d_out, int out_size, void* d_ws, size_t ws_size,
                              hipStream_t stream) {
  (void)in_sizes; (void)n_in; (void)out_size; (void)ws_size;

  const float* x            = (const float*)d_in[0];
  const float* cached_key   = (const float*)d_in[1];
  const float* cached_val   = (const float*)d_in[2];
  const float* pos_emb      = (const float*)d_in[3];
  const float* in_proj_w    = (const float*)d_in[4];
  const float* in_proj_b    = (const float*)d_in[5];
  const float* linear_pos_w = (const float*)d_in[6];
  const float* out_proj_w   = (const float*)d_in[7];
  const float* out_proj_b   = (const float*)d_in[8];

  float* out     = (float*)d_out;                        // [S,B,E]
  float* weights = out + (size_t)kS * kB * kE;           // [B*H,S,KV]
  float* cko     = weights + (size_t)kB * kH * kS * kKV; // [S,B,A]
  float* cvo     = cko + (size_t)kS * kB * kA;           // [S,B,A/2]

  auto alignup = [](size_t x) { return (x + 255) & ~(size_t)255; };
  char* ws = (char*)d_ws;
  float* xp      = (float*)ws; ws += alignup(sizeof(float) * (size_t)kS * kB * kIO);
  float* attnout = (float*)ws; ws += alignup(sizeof(float) * (size_t)kS * kB * (kA / 2));
  float* posm    = (float*)ws; ws += alignup(sizeof(float) * (size_t)kS2 * kH * kP);
  float* Pp      = (float*)ws; ws += alignup(sizeof(float) * (size_t)kB * kH * kS * kP);
  __bf16* Qb     = (__bf16*)ws; ws += alignup(sizeof(__bf16) * (size_t)kB * kH * kS * kHD);
  __bf16* Kb     = (__bf16*)ws; ws += alignup(sizeof(__bf16) * (size_t)kB * kH * kKV * kHD);
  __bf16* Vt     = (__bf16*)ws; ws += alignup(sizeof(__bf16) * (size_t)kB * kH * kKV * kVD);

  const int SB = kS * kB;  // 8192

  // 1) xp = x @ in_proj_w^T + b        [8192 x 1312 x 512]
  {
    dim3 g((kIO + 63) / 64, SB / 128);
    gemm_bf16_wmma<<<g, 256, 0, stream>>>(x, in_proj_w, in_proj_b, xp, SB, kIO, kE);
  }
  // 2) posm = pos_emb @ linear_pos_w^T [1535 x 32 x 512]
  {
    dim3 g(1, (kS2 + 127) / 128);
    gemm_bf16_wmma<<<g, 256, 0, stream>>>(pos_emb, linear_pos_w, nullptr, posm,
                                          kS2, kH * kP, kE);
  }
  // 3) pack Q/K/V (bf16), P (fp32) + cache outputs
  {
    size_t nq4 = (size_t)kB * kH * kS * kHD / 4;
    pack_q_kernel<<<(unsigned)((nq4 + 255) / 256), 256, 0, stream>>>(xp, Qb);
    size_t nk4 = (size_t)kB * kH * kKV * kHD / 4;
    pack_k_kernel<<<(unsigned)((nk4 + 255) / 256), 256, 0, stream>>>(xp, cached_key, Kb);
    dim3 gv(kKV / 64, kH, kB);
    pack_v_kernel<<<gv, 256, 0, stream>>>(xp, cached_val, Vt);
    size_t np = (size_t)kB * kH * kS;
    pack_p_kernel<<<(unsigned)((np + 255) / 256), 256, 0, stream>>>(xp, Pp);
    size_t nc4 = (size_t)kS * kB * (kA + kA / 2) / 4;
    cache_out_kernel<<<(unsigned)((nc4 + 255) / 256), 256, 0, stream>>>(xp, cko, cvo);
  }
  // 4) attention (scores + rel-pos gather + softmax + AV), writes weights
  {
    dim3 g(kS / 16, kH, kB);
    attn_kernel<<<g, 256, 0, stream>>>(Qb, Kb, Vt, Pp, posm, weights, attnout);
  }
  // 5) out = attnout @ out_proj_w^T + b [8192 x 512 x 256]
  {
    dim3 g(kE / 64, SB / 128);
    gemm_bf16_wmma<<<g, 256, 0, stream>>>(attnout, out_proj_w, out_proj_b, out,
                                          SB, kE, kA / 2);
  }
}
